// MultiHeadCrossAttention_23794118819991
// MI455X (gfx1250) — matrix operand order repaired and checked by reading, not compiled, not run on hardware
//
#include <hip/hip_runtime.h>
#include <hip/hip_bf16.h>

// ---------------------------------------------------------------------------
// MultiHeadCrossAttention for MI455X (gfx1250, wave32, WMMA).
// All contractions via v_wmma_f32_16x16x32_bf16 (f32 accumulate).
// Async tensor-to-LDS staging via global_load_async_to_lds_b128 (ASYNCcnt).
// ---------------------------------------------------------------------------

typedef __attribute__((ext_vector_type(16))) __bf16 v16bf;
typedef __attribute__((ext_vector_type(8)))  __bf16 v8bf;
typedef __attribute__((ext_vector_type(8)))  float  v8f;

__device__ __forceinline__ v16bf cat8(v8bf lo, v8bf hi) {
    return __builtin_shufflevector(lo, hi, 0,1,2,3,4,5,6,7,8,9,10,11,12,13,14,15);
}

__device__ __forceinline__ v8f wmma_bf16(v16bf a, v16bf b, v8f c) {
    // (neg_a, A, neg_b, B, c_mod, C, reuse_a, reuse_b)
    return __builtin_amdgcn_wmma_f32_16x16x32_bf16(false, a, false, b, (short)0, c, false, false);
}

__device__ __forceinline__ unsigned pack_bf16(float a, float b) {
    union { __bf16 h; unsigned short u; } ca, cb;
    ca.h = (__bf16)a; cb.h = (__bf16)b;
    return (unsigned)ca.u | ((unsigned)cb.u << 16);
}

// Async global->LDS 128-bit copy (GV mode, ASYNCcnt-tracked).
// vdst = LDS byte offset VGPR, vaddr = 64-bit global address VGPR pair.
__device__ __forceinline__ void async_load_b128(const void* gaddr, void* lds) {
    unsigned lo = (unsigned)(unsigned long long)lds;       // addrspace(3) offset
    unsigned long long ga = (unsigned long long)gaddr;
    asm volatile("global_load_async_to_lds_b128 %0, %1, off"
                 :: "v"(lo), "v"(ga) : "memory");
}
__device__ __forceinline__ void wait_async() {
    asm volatile("s_wait_asynccnt 0x0" ::: "memory");
}

// ---------------------------------------------------------------------------
// One-time weight prep: W f32 [K,N] row-major -> Wt bf16 [N,K] row-major.
// ---------------------------------------------------------------------------
__global__ __launch_bounds__(256)
void transpose_to_bf16_kernel(const float* __restrict__ W, __bf16* __restrict__ Wt,
                              int K, int N) {
    __shared__ float tile[32][33];
    const int k0 = blockIdx.y * 32, n0 = blockIdx.x * 32;
    const int tx = threadIdx.x & 31, ty = threadIdx.x >> 5;   // 32 x 8
    #pragma unroll
    for (int j = 0; j < 4; ++j)
        tile[ty + j * 8][tx] = W[(size_t)(k0 + ty + j * 8) * N + n0 + tx];
    __syncthreads();
    #pragma unroll
    for (int j = 0; j < 4; ++j)
        Wt[(size_t)(n0 + ty + j * 8) * K + k0 + tx] = (__bf16)tile[tx][ty + j * 8];
}

// ---------------------------------------------------------------------------
// GEMM: C[M,N] = A[M,K](f32) * W[K,N] (+ bias); W pre-transposed bf16 Wt[N,K].
// Block tile 128x128, BK=32, 8 waves; wave owns 32x64 = 2x4 WMMA tiles.
// B tile staged with async-to-LDS; A tile converted f32->bf16 (packed b64).
// ---------------------------------------------------------------------------
template<bool OUT_BF16, bool BIAS>
__global__ __launch_bounds__(256)
void gemm_wmma_kernel(const float* __restrict__ A, const __bf16* __restrict__ Wt,
                      const float* __restrict__ bias, void* __restrict__ Cout,
                      int M, int N, int K) {
    __shared__ __bf16 As[128][32];   // [m][k]
    __shared__ __bf16 Bs[128][32];   // [n][k]
    const int tid  = threadIdx.x;
    const int row0 = blockIdx.y * 128;
    const int col0 = blockIdx.x * 128;
    const int w  = tid >> 5, l = tid & 31;
    const int wm = w & 3,  wn = w >> 2;
    const int lm = l & 15, hi = l >> 4;
    const int c0 = hi ? 16 : 0;

    v8f acc[2][4] = {};

    for (int k0 = 0; k0 < K; k0 += 32) {
        // --- B tile: async global->LDS (pure bf16 copy) ---
        {
            int n = tid >> 1, kk = (tid & 1) * 16;
            const __bf16* src = Wt + (size_t)(col0 + n) * K + k0 + kk;
            async_load_b128(src,     &Bs[n][kk]);
            async_load_b128(src + 8, &Bs[n][kk + 8]);
        }
        // --- A tile: f32 -> packed bf16, b64 LDS stores ---
        #pragma unroll
        for (int j = 0; j < 4; ++j) {
            int e = (tid + j * 256) * 4;
            int r = e >> 5, c = e & 31;
            const float4 f = *(const float4*)(A + (size_t)(row0 + r) * K + k0 + c);
            uint2 p; p.x = pack_bf16(f.x, f.y); p.y = pack_bf16(f.z, f.w);
            *(uint2*)&As[r][c] = p;
        }
        // --- prefetch next K-step A tile ---
        if (k0 + 32 < K) {
            int e = tid * 4;
            __builtin_prefetch(A + (size_t)(row0 + (e >> 5)) * K + k0 + 32 + (e & 31), 0, 1);
        }
        wait_async();
        __syncthreads();

        // --- fragments (all loads batched before the WMMA chain) ---
        v16bf afrag[2], bfrag[4];
        #pragma unroll
        for (int rg = 0; rg < 2; ++rg) {
            int r = wm * 32 + rg * 16 + lm;
            afrag[rg] = cat8(*(const v8bf*)&As[r][c0], *(const v8bf*)&As[r][c0 + 8]);
        }
        #pragma unroll
        for (int cg = 0; cg < 4; ++cg) {
            int c = wn * 64 + cg * 16 + lm;
            bfrag[cg] = *(const v16bf*)&Bs[c][hi ? 16 : 0];
        }
        #pragma unroll
        for (int rg = 0; rg < 2; ++rg)
            #pragma unroll
            for (int cg = 0; cg < 4; ++cg)
                acc[rg][cg] = wmma_bf16(afrag[rg], bfrag[cg], acc[rg][cg]);
        __syncthreads();
    }

    #pragma unroll
    for (int rg = 0; rg < 2; ++rg) {
        #pragma unroll
        for (int cg = 0; cg < 4; ++cg) {
            int col = col0 + wn * 64 + cg * 16 + lm;
            float bv = BIAS ? bias[col] : 0.0f;
            #pragma unroll
            for (int i = 0; i < 8; ++i) {
                int r = row0 + wm * 32 + rg * 16 + i + (hi ? 8 : 0);
                float v = acc[rg][cg][i] + bv;
                if (OUT_BF16)
                    ((__bf16*)Cout)[(size_t)r * N + col] = (__bf16)v;
                else
                    ((float*)Cout)[(size_t)r * N + col] = v;
            }
        }
    }
}

// ---------------------------------------------------------------------------
// RoPE in-place on Q and K half of KV; the attention scale 1/sqrt(64) is
// folded into Q here (removes a per-tile VALU pass in the flash kernel).
// ---------------------------------------------------------------------------
__global__ __launch_bounds__(256)
void rope_kernel(__bf16* __restrict__ q, __bf16* __restrict__ kv, int BN, int N) {
    int unit = blockIdx.x * 8 + (threadIdx.x >> 5);
    int j    = threadIdx.x & 31;
    int h    = unit & 15;
    int row  = unit >> 4;
    if (row >= BN) return;
    int n = row & (N - 1);

    float inv = __expf(-(float)j * 0.28782313662425f);   // ln(10000)/32
    float ang = (float)n * inv;
    float cs  = __cosf(ang);
    float sn  = __sinf(ang);
    const float qs = 0.125f;                             // 64^-0.5

    __bf16* qp = q + (size_t)row * 1024 + h * 64;
    float x1 = (float)qp[j], x2 = (float)qp[j + 32];
    qp[j]      = (__bf16)((x1 * cs - x2 * sn) * qs);
    qp[j + 32] = (__bf16)((x2 * cs + x1 * sn) * qs);

    __bf16* kp = kv + (size_t)row * 2048 + h * 64;
    x1 = (float)kp[j]; x2 = (float)kp[j + 32];
    kp[j]      = (__bf16)(x1 * cs - x2 * sn);
    kp[j + 32] = (__bf16)(x2 * cs + x1 * sn);
}

// ---------------------------------------------------------------------------
// Flash attention: workgroup = (b,h) x 64 query rows (4 waves x 16 rows),
// kv tile = 64. K tile async-to-LDS; V tile transposed on store.
// Row sums via P @ ones on the matrix unit; row max via half-wave butterfly.
// ---------------------------------------------------------------------------
__global__ __launch_bounds__(128)
void flash_attn_kernel(const __bf16* __restrict__ q, const __bf16* __restrict__ kv,
                       float* __restrict__ out, int Nq, int Nkv) {
    __shared__ __bf16 Ks[64][64];        // [kv][d]
    __shared__ __bf16 Vs[64][64];        // [d][kv]
    __shared__ __bf16 Ps[4][16][64];     // per-wave P staging

    const int bh = blockIdx.y;
    const int b  = bh >> 4, h = bh & 15;
    const int w  = threadIdx.x >> 5, l = threadIdx.x & 31;
    const int lm = l & 15, hi = l >> 4;
    const int c0  = hi ? 16 : 0;         // A-frag element split
    const int kof = hi ? 16 : 0;         // B-frag K split
    const int qbase = blockIdx.x * 64 + w * 16;

    const __bf16* qp = q + ((size_t)(b * Nq + qbase + lm)) * 1024 + h * 64;
    v16bf aq0 = cat8(*(const v8bf*)(qp + c0),      *(const v8bf*)(qp + c0 + 8));
    v16bf aq1 = cat8(*(const v8bf*)(qp + 32 + c0), *(const v8bf*)(qp + 32 + c0 + 8));

    v16bf onesb;
    #pragma unroll
    for (int e = 0; e < 16; ++e) onesb[e] = (__bf16)1.0f;

    float m_i[8], l_i[8];
    v8f acc[4] = {};
    #pragma unroll
    for (int i = 0; i < 8; ++i) { m_i[i] = -1e30f; l_i[i] = 0.0f; }

    const int tid = threadIdx.x;
    const int lr  = tid >> 1;            // 0..63: kv row
    const int ld0 = (tid & 1) * 32;      // d half

    for (int kt = 0; kt < Nkv; kt += 64) {
        // --- K tile: async global->LDS, 4 x b128 per thread ---
        {
            const __bf16* src = kv + ((size_t)(b * Nkv + kt + lr)) * 2048 + h * 64 + ld0;
            #pragma unroll
            for (int u = 0; u < 4; ++u)
                async_load_b128(src + u * 8, &Ks[lr][ld0 + u * 8]);
        }
        // --- V tile: transpose on store ---
        {
            const __bf16* src = kv + ((size_t)(b * Nkv + kt + lr)) * 2048 + 1024 + h * 64 + ld0;
            #pragma unroll
            for (int u = 0; u < 4; ++u) {
                v8bf vv = *(const v8bf*)(src + u * 8);
                #pragma unroll
                for (int jj = 0; jj < 8; ++jj)
                    Vs[ld0 + u * 8 + jj][lr] = vv[jj];
            }
        }
        if (kt + 64 < Nkv) {
            __builtin_prefetch(kv + ((size_t)(b * Nkv + kt + 64 + lr)) * 2048 + h * 64 + ld0, 0, 1);
            __builtin_prefetch(kv + ((size_t)(b * Nkv + kt + 64 + lr)) * 2048 + 1024 + h * 64 + ld0, 0, 1);
        }
        wait_async();
        __syncthreads();

        // --- S = Q K^T : 4 kv col-chunks (Q pre-scaled by 1/8 in RoPE) ---
        v8f s[4];
        #pragma unroll
        for (int nc = 0; nc < 4; ++nc) {
            v16bf bk0 = *(const v16bf*)&Ks[nc * 16 + lm][kof];
            v16bf bk1 = *(const v16bf*)&Ks[nc * 16 + lm][32 + kof];
            v8f t = {};
            t = wmma_bf16(aq0, bk0, t);
            t = wmma_bf16(aq1, bk1, t);
            s[nc] = t;
        }

        // --- online softmax: max via half-wave butterfly ---
        v8f p[4];
        float alpha[8];
        #pragma unroll
        for (int i = 0; i < 8; ++i) {
            float rm = fmaxf(fmaxf(s[0][i], s[1][i]), fmaxf(s[2][i], s[3][i]));
            #pragma unroll
            for (int off = 1; off < 16; off <<= 1)
                rm = fmaxf(rm, __shfl_xor(rm, off, 32));
            float mn = fmaxf(m_i[i], rm);
            alpha[i] = __expf(m_i[i] - mn);
            m_i[i]   = mn;
            #pragma unroll
            for (int nc = 0; nc < 4; ++nc)
                p[nc][i] = __expf(s[nc][i] - mn);
        }

        // --- P: C-layout -> bf16 A-layout via per-wave LDS staging ---
        #pragma unroll
        for (int i = 0; i < 8; ++i) {
            int r = i + (hi ? 8 : 0);
            #pragma unroll
            for (int nc = 0; nc < 4; ++nc)
                Ps[w][r][nc * 16 + lm] = (__bf16)p[nc][i];
        }
        v16bf ap0 = cat8(*(const v8bf*)&Ps[w][lm][c0],      *(const v8bf*)&Ps[w][lm][c0 + 8]);
        v16bf ap1 = cat8(*(const v8bf*)&Ps[w][lm][32 + c0], *(const v8bf*)&Ps[w][lm][32 + c0 + 8]);

        // --- row sums on the matrix unit ---
        v8f lsum = {};
        lsum = wmma_bf16(ap0, onesb, lsum);
        lsum = wmma_bf16(ap1, onesb, lsum);

        #pragma unroll
        for (int i = 0; i < 8; ++i)
            l_i[i] = l_i[i] * alpha[i] + lsum[i];
        #pragma unroll
        for (int jj = 0; jj < 4; ++jj)
            #pragma unroll
            for (int i = 0; i < 8; ++i)
                acc[jj][i] *= alpha[i];

        // --- O += P V : 8 WMMAs over d chunks x kv halves ---
        #pragma unroll
        for (int jj = 0; jj < 4; ++jj) {
            v16bf bv0 = *(const v16bf*)&Vs[jj * 16 + lm][kof];
            v16bf bv1 = *(const v16bf*)&Vs[jj * 16 + lm][32 + kof];
            acc[jj] = wmma_bf16(ap0, bv0, acc[jj]);
            acc[jj] = wmma_bf16(ap1, bv1, acc[jj]);
        }
        __syncthreads();
    }

    // --- normalize and store as [B, Nq, H*64] f32 ---
    #pragma unroll
    for (int jj = 0; jj < 4; ++jj) {
        #pragma unroll
        for (int i = 0; i < 8; ++i) {
            int r = qbase + i + (hi ? 8 : 0);
            int d = jj * 16 + lm;
            out[((size_t)(b * Nq + r)) * 1024 + h * 64 + d] = acc[jj][i] / l_i[i];
        }
    }
}

// ---------------------------------------------------------------------------
// Launcher
// ---------------------------------------------------------------------------
extern "C" void kernel_launch(void* const* d_in, const int* in_sizes, int n_in,
                              void* d_out, int out_size, void* d_ws, size_t ws_size,
                              hipStream_t stream) {
    const float* x   = (const float*)d_in[0];
    const float* ctx = (const float*)d_in[1];
    const float* Wq  = (const float*)d_in[2];
    const float* Wkv = (const float*)d_in[3];
    const float* Wo  = (const float*)d_in[4];
    const float* bo  = (const float*)d_in[5];

    const int B = 2, N = 2048, C = 1024;
    const int M = B * N;

    char* ws = (char*)d_ws;
    size_t off = 0;
    __bf16* qbuf  = (__bf16*)(ws + off); off += (size_t)M * C * 2;
    __bf16* kvbuf = (__bf16*)(ws + off); off += (size_t)M * 2 * C * 2;
    float*  abuf  = (float*)(ws + off);  off += (size_t)M * C * 4;
    __bf16* Wqt   = (__bf16*)(ws + off); off += (size_t)C * C * 2;
    __bf16* Wkvt  = (__bf16*)(ws + off); off += (size_t)2 * C * C * 2;
    __bf16* Wot   = (__bf16*)(ws + off);

    transpose_to_bf16_kernel<<<dim3(C / 32, C / 32), 256, 0, stream>>>(Wq, Wqt, C, C);
    transpose_to_bf16_kernel<<<dim3(2 * C / 32, C / 32), 256, 0, stream>>>(Wkv, Wkvt, C, 2 * C);
    transpose_to_bf16_kernel<<<dim3(C / 32, C / 32), 256, 0, stream>>>(Wo, Wot, C, C);

    gemm_wmma_kernel<true, false><<<dim3(C / 128, M / 128), 256, 0, stream>>>(
        x, Wqt, nullptr, qbuf, M, C, C);
    gemm_wmma_kernel<true, false><<<dim3(2 * C / 128, M / 128), 256, 0, stream>>>(
        ctx, Wkvt, nullptr, kvbuf, M, 2 * C, C);
    rope_kernel<<<(M * 16) / 8, 256, 0, stream>>>(qbuf, kvbuf, M, N);
    flash_attn_kernel<<<dim3(N / 64, B * 16), 128, 0, stream>>>(
        qbuf, kvbuf, abuf, N, N);
    gemm_wmma_kernel<false, true><<<dim3(C / 128, M / 128), 256, 0, stream>>>(
        abuf, Wot, bo, (float*)d_out, M, C, C);
}
